// Social_Model_52828097741379
// MI455X (gfx1250) — matrix operand order repaired and checked by reading, not compiled, not run on hardware
//
#include <hip/hip_runtime.h>

typedef __attribute__((ext_vector_type(16))) _Float16 v16h;
typedef __attribute__((ext_vector_type(8)))  _Float16 v8h;
typedef __attribute__((ext_vector_type(8)))  float    v8f;
typedef __attribute__((ext_vector_type(2)))  float    v2f;

#define B_    512
#define NN    32
#define TT    50
#define HENC  64
#define GENC  256
#define HDEC  128
#define GDEC  512
#define TDEC  30

__device__ __forceinline__ v16h cat8(v8h lo, v8h hi) {
  return __builtin_shufflevector(lo, hi, 0,1,2,3,4,5,6,7,8,9,10,11,12,13,14,15);
}
// branchless activations: exp pipe + raw v_rcp, no ocml branches
__device__ __forceinline__ float sigmf(float x){
  return __builtin_amdgcn_rcpf(1.0f + __expf(-x));
}
__device__ __forceinline__ float tanh_fast(float x){
  return __builtin_fmaf(2.0f, __builtin_amdgcn_rcpf(1.0f + __expf(-2.0f * x)), -1.0f);
}
__device__ __forceinline__ v8f wmma_f16(v16h a, v16h b, v8f c){
  return __builtin_amdgcn_wmma_f32_16x16x32_f16(false, a, false, b, (short)0, c, false, false);
}
// 16x16x4 f32 WMMA: carries the (x0,x1,bias) rank-3 update exactly in f32
__device__ __forceinline__ v8f wmma_f32k4(v2f a, v2f b, v8f c){
  return __builtin_amdgcn_wmma_f32_16x16x4_f32(false, a, false, b, (short)0, c, false, false);
}

// ---------------- prep: weight conversion ----------------
__global__ __launch_bounds__(256) void prep_kernel(
    const float* __restrict__ a_whh, const float* __restrict__ n_whh,
    const float* __restrict__ d_wih, const float* __restrict__ d_whh,
    const float* __restrict__ d_bih, const float* __restrict__ d_bhh,
    _Float16* __restrict__ nW, _Float16* __restrict__ aW,
    _Float16* __restrict__ dW, float* __restrict__ dbsum)
{
  int i = blockIdx.x * 256 + threadIdx.x;
  if (i < GDEC * HDEC) dW[i] = (_Float16)(d_wih[i] + d_whh[i]);
  if (i < GDEC) dbsum[i] = d_bih[i] + d_bhh[i];
  if (i < GENC * HENC){ nW[i] = (_Float16)n_whh[i]; aW[i] = (_Float16)a_whh[i]; }
}

// ---------------- fused agent+neighbour LSTM encoder ----------------
// one wave = 16 sequences, full 256-gate width, 50 timesteps.
// Recurrent weights live in VGPRs for the whole T loop.
#define HSTRIDE 72   // padded LDS row stride (halfs) to spread banks
__global__ __launch_bounds__(128) void encoder_kernel(
    const float* __restrict__ agent_x, const float* __restrict__ neigh_x,
    const _Float16* __restrict__ nW, const float* __restrict__ n_wih,
    const float* __restrict__ n_bih, const float* __restrict__ n_bhh,
    const _Float16* __restrict__ aW, const float* __restrict__ a_wih,
    const float* __restrict__ a_bih, const float* __restrict__ a_bhh,
    float* __restrict__ nh, float* __restrict__ ah)
{
  __shared__ __align__(16) _Float16 hbuf[4][16 * HSTRIDE];
  const int tid  = threadIdx.x;
  const int wave = tid >> 5, lane = tid & 31;
  const int nlo  = lane & 15, hi = (lane >> 4) & 1;
  const int task = blockIdx.x * 4 + wave;
  const bool isAgent = task >= 1024;
  const float*     xb   = isAgent ? agent_x : neigh_x;
  const _Float16*  W    = isAgent ? aW : nW;
  const float*     wih  = isAgent ? a_wih : n_wih;
  const float*     bih  = isAgent ? a_bih : n_bih;
  const float*     bhh  = isAgent ? a_bhh : n_bhh;
  float*           outp = isAgent ? ah : nh;
  const int m0 = (isAgent ? (task - 1024) : task) * 16;

  _Float16* hl = &hbuf[wave][0];
  { // zero the h tile (h0 = 0)
    unsigned* hz = (unsigned*)hl;
    #pragma unroll
    for (int i = 0; i < (16*HSTRIDE/2); i += 32) hz[i + lane] = 0u;
  }

  // ---- loop-invariant B operands, register-resident ----
  v16h Bh[32];        // 16 tiles x 2 K-chunks of recurrent weights (f16)
  v2f  Bxv[16];       // per-tile (wih0,wih1 | bias,0) for the f32 K=4 WMMA
  #pragma unroll
  for (int tile = 0; tile < 16; ++tile){
    const int n = tile * 16 + nlo;
    const _Float16* wp = W + n * HENC + 16 * hi;   // B layout: 16 contiguous K per lane
    Bh[tile*2+0] = *(const v16h*)(wp);
    Bh[tile*2+1] = *(const v16h*)(wp + 32);
    float bs = bih[n] + bhh[n];
    v2f bx;
    bx[0] = (lane < 16) ? wih[2*n+0] : bs;         // K0 row / K2 row
    bx[1] = (lane < 16) ? wih[2*n+1] : 0.0f;       // K1 row / K3 row
    Bxv[tile] = bx;
  }

  v8f zero8 = {};
  v8f cst[4];
  #pragma unroll
  for (int j = 0; j < 4; ++j) cst[j] = zero8;

  const _Float16* rowp = hl + nlo * HSTRIDE;          // A-layout: M = lane%16
  const float*    xrow = xb + (size_t)(m0 + nlo) * (TT*2);

  for (int t = 0; t < TT; ++t){
    // previous step's cross-lane h stores must land before A-layout reads
    asm volatile("s_wait_dscnt 0x0" ::: "memory");
    v16h a0 = cat8(*(const v8h*)(rowp +      16*hi), *(const v8h*)(rowp + 8 + 16*hi));
    v16h a1 = cat8(*(const v8h*)(rowp + 32 + 16*hi), *(const v8h*)(rowp + 40 + 16*hi));
    // f32 A chunk: rows (x0,x1) for K0/K1 on lanes 0-15, (1,0) for K2/K3 on lanes 16-31
    float x0 = xrow[t*2 + 0], x1 = xrow[t*2 + 1];
    v2f axf;
    axf[0] = (lane < 16) ? x0 : 1.0f;
    axf[1] = (lane < 16) ? x1 : 0.0f;

    #pragma unroll
    for (int j = 0; j < 4; ++j){            // 16-col slice of hidden dim
      v8f g4[4];
      #pragma unroll
      for (int gg = 0; gg < 4; ++gg){       // gate groups i,f,g,o
        const int tile = gg * 4 + j;
        v8f acc = wmma_f32k4(axf, Bxv[tile], zero8);   // x·wih + bias (exact f32)
        acc = wmma_f16(a0, Bh[2*tile+0], acc);
        acc = wmma_f16(a1, Bh[2*tile+1], acc);
        g4[gg] = acc;
      }
      v8f c = cst[j];
      const int n = j * 16 + nlo;
      #pragma unroll
      for (int r = 0; r < 8; ++r){
        float ig = sigmf(g4[0][r]);
        float fg = sigmf(g4[1][r]);
        float gv = tanh_fast(g4[2][r]);
        float og = sigmf(g4[3][r]);
        float cn = fg * c[r] + ig * gv;
        c[r]  = cn;
        float hn = og * tanh_fast(cn);
        hl[(r + 8*hi) * HSTRIDE + n] = (_Float16)hn;   // C-layout -> [M][k] f16
      }
      cst[j] = c;
    }
  }
  // final h: read back the f16 tile (exactly the values the recurrence consumed)
  asm volatile("s_wait_dscnt 0x0" ::: "memory");
  #pragma unroll
  for (int j = 0; j < 4; ++j){
    const int n = j * 16 + nlo;
    #pragma unroll
    for (int r = 0; r < 8; ++r)
      outp[(size_t)(m0 + r + 8*hi) * HENC + n] = (float)hl[(r + 8*hi) * HSTRIDE + n];
  }
}

// ---------------- mean over neighbours + concat ----------------
__global__ __launch_bounds__(256) void combine_kernel(
    const float* __restrict__ ah, const float* __restrict__ nh, float* __restrict__ h0)
{
  int i = blockIdx.x * 256 + threadIdx.x;
  if (i >= B_ * HDEC) return;
  int b = i >> 7, j = i & 127;
  float v;
  if (j < HENC) v = ah[b * HENC + j];
  else {
    int n = j - HENC;
    float s = 0.f;
    #pragma unroll 8
    for (int k = 0; k < NN; ++k) s += nh[(size_t)(b * NN + k) * HENC + n];
    v = s * (1.0f / NN);
  }
  h0[i] = v;
}

// ---------------- decoder LSTMCell (input == h) ----------------
#define DSTRIDE 136
__global__ __launch_bounds__(128) void decoder_kernel(
    const float* __restrict__ h0, const _Float16* __restrict__ dW,
    const float* __restrict__ db,
    const float* __restrict__ e_w, const float* __restrict__ e_b,
    float* __restrict__ out)
{
  __shared__ __align__(16) _Float16 hs16[16 * DSTRIDE];
  __shared__ __align__(16) float    hs32[16 * HDEC];
  const int tid  = threadIdx.x;
  const int wave = tid >> 5, lane = tid & 31;
  const int nlo  = lane & 15, hi = (lane >> 4) & 1;
  const int m0 = blockIdx.x * 16;

  for (int i = tid; i < 16 * HDEC; i += 128){
    int r = i >> 7, k = i & 127;
    float v = h0[(size_t)(m0 + r) * HDEC + k];
    hs32[r * HDEC + k]    = v;
    hs16[r * DSTRIDE + k] = (_Float16)v;
  }
  // loop-invariant B: this wave's 8 gate tiles x 4 K-chunks, register-resident
  v16h Bd[32];
  float biasv[2][4];
  #pragma unroll
  for (int gg = 0; gg < 4; ++gg)
    #pragma unroll
    for (int j = 0; j < 2; ++j){
      const int tile = gg*8 + wave*2 + j;
      const int n = tile * 16 + nlo;
      const _Float16* wp = dW + (size_t)n * HDEC + 16*hi;
      #pragma unroll
      for (int c = 0; c < 4; ++c)
        Bd[(gg*2+j)*4 + c] = *(const v16h*)(wp + 32*c);
      biasv[j][gg] = db[n];
    }

  v8f zero8 = {};
  v8f cst[2]; cst[0] = zero8; cst[1] = zero8;
  const _Float16* rowp = hs16 + nlo * DSTRIDE;
  __syncthreads();

  for (int t = 0; t < TDEC; ++t){
    v16h a[4];
    #pragma unroll
    for (int c = 0; c < 4; ++c)
      a[c] = cat8(*(const v8h*)(rowp + 32*c + 16*hi), *(const v8h*)(rowp + 32*c + 8 + 16*hi));
    __syncthreads();                      // all waves done reading h_{t-1}
    #pragma unroll
    for (int j = 0; j < 2; ++j){
      v8f g4[4];
      #pragma unroll
      for (int gg = 0; gg < 4; ++gg){
        v8f acc;
        #pragma unroll
        for (int r = 0; r < 8; ++r) acc[r] = biasv[j][gg];
        #pragma unroll
        for (int c = 0; c < 4; ++c)
          acc = wmma_f16(a[c], Bd[(gg*2+j)*4 + c], acc);
        g4[gg] = acc;
      }
      v8f c2 = cst[j];
      const int nh_ = wave*32 + j*16 + nlo;
      #pragma unroll
      for (int r = 0; r < 8; ++r){
        float ig = sigmf(g4[0][r]);
        float fg = sigmf(g4[1][r]);
        float gv = tanh_fast(g4[2][r]);
        float og = sigmf(g4[3][r]);
        float cn = fg * c2[r] + ig * gv;
        c2[r] = cn;
        float hn = og * tanh_fast(cn);
        hs16[(r + 8*hi) * DSTRIDE + nh_] = (_Float16)hn;
        hs32[(r + 8*hi) * HDEC    + nh_] = hn;
      }
      cst[j] = c2;
    }
    __syncthreads();                      // h_t visible to everyone
    if (tid < 32){                        // prediction head: 16 rows x 2 dims
      int r = tid >> 1, d = tid & 1;
      float s = e_b[d];
      const float* hr = hs32 + r * HDEC;
      const float* wr = e_w + d * HDEC;
      #pragma unroll 8
      for (int k = 0; k < HDEC; ++k) s += hr[k] * wr[k];
      out[(size_t)(m0 + r) * (TDEC*2) + t*2 + d] = s;
    }
  }
}

extern "C" void kernel_launch(void* const* d_in, const int* in_sizes, int n_in,
                              void* d_out, int out_size, void* d_ws, size_t ws_size,
                              hipStream_t stream)
{
  (void)in_sizes; (void)n_in; (void)out_size; (void)ws_size;
  const float* agent_traj = (const float*)d_in[0];
  const float* neigh_traj = (const float*)d_in[1];
  const float* a_wih = (const float*)d_in[2];
  const float* a_whh = (const float*)d_in[3];
  const float* a_bih = (const float*)d_in[4];
  const float* a_bhh = (const float*)d_in[5];
  const float* n_wih = (const float*)d_in[6];
  const float* n_whh = (const float*)d_in[7];
  const float* n_bih = (const float*)d_in[8];
  const float* n_bhh = (const float*)d_in[9];
  const float* d_wih = (const float*)d_in[10];
  const float* d_whh = (const float*)d_in[11];
  const float* d_bih = (const float*)d_in[12];
  const float* d_bhh = (const float*)d_in[13];
  const float* e_w   = (const float*)d_in[14];
  const float* e_b   = (const float*)d_in[15];

  char* ws = (char*)d_ws;
  _Float16* nW    = (_Float16*)(ws + 0);         // 32768
  _Float16* aW    = (_Float16*)(ws + 32768);     // 32768
  _Float16* dW    = (_Float16*)(ws + 65536);     // 131072
  float*    dbsum = (float*)   (ws + 196608);    // 2048
  float*    ah    = (float*)   (ws + 198656);    // 512*64*4   = 131072
  float*    nh    = (float*)   (ws + 329728);    // 16384*64*4 = 4194304
  float*    h0    = (float*)   (ws + 4524032);   // 512*128*4  = 262144

  prep_kernel<<<256, 256, 0, stream>>>(a_whh, n_whh, d_wih, d_whh, d_bih, d_bhh,
                                       nW, aW, dW, dbsum);
  // 1024 neighbour wave-tasks + 32 agent wave-tasks, 4 waves per block
  encoder_kernel<<<264, 128, 0, stream>>>(agent_traj, neigh_traj,
                                          nW, n_wih, n_bih, n_bhh,
                                          aW, a_wih, a_bih, a_bhh, nh, ah);
  combine_kernel<<<256, 256, 0, stream>>>(ah, nh, h0);
  decoder_kernel<<<32, 128, 0, stream>>>(h0, dW, dbsum, e_w, e_b, (float*)d_out);
}
